// MultiHeadMambaPredictor_34471407517777
// MI455X (gfx1250) — matrix operand order repaired and checked by reading, not compiled, not run on hardware
//
#include <hip/hip_runtime.h>
#include <hip/hip_bf16.h>

typedef __attribute__((ext_vector_type(16))) _Float16 v16h;
typedef __attribute__((ext_vector_type(8)))  float    v8f;

#define B_  32
#define L_  1024
#define F_  20
#define H_  256
#define N_  64
#define KC_ 4
#define E_  512
#define R_  16
#define M_  (B_*L_)          // 32768 tokens
#define DTB_ (R_+N_)          // 80 cols stored from x_proj

// ---------------------------------------------------------------------------
// 1) Convert in_proj_W rows [0,512) (the xin half) to f16 for WMMA B-frags.
// ---------------------------------------------------------------------------
__global__ void k_w16(const float* __restrict__ W, _Float16* __restrict__ W16) {
  int i = blockIdx.x * 256 + threadIdx.x;          // 512*256 elems
  if (i < E_ * H_) W16[i] = (_Float16)W[i];
}

// ---------------------------------------------------------------------------
// 2) Fused embed + in_proj(xin half) GEMM via V_WMMA_F32_16X16X32_F16.
//    One wave (32 thr) per (M-tile of 16 rows, N-group of 128 cols).
//    h0 tile (16x256) built in LDS as f16; B frags read from f16 weights.
// ---------------------------------------------------------------------------
__global__ __launch_bounds__(32) void k_embed_inproj(
    const float* __restrict__ x, const float* __restrict__ embW,
    const float* __restrict__ embB, const _Float16* __restrict__ W16,
    float* __restrict__ xin)
{
  __shared__ float     xs[16 * F_];
  __shared__ _Float16  h0s[16 * H_];
  const int lane  = threadIdx.x;
  const int tileM = blockIdx.x >> 2;     // 0..2047
  const int ngrp  = blockIdx.x & 3;      // 0..3 -> N base = ngrp*128
  const int m0    = tileM * 16;

  for (int i = lane; i < 16 * F_; i += 32) {
    int r = i / F_, f = i % F_;
    xs[i] = x[(size_t)(m0 + r) * F_ + f];
  }
  __syncthreads();

  // each lane computes 8 columns x 16 rows of h0 = x @ embed_W.T + b
  for (int j = 0; j < H_ / 32; ++j) {
    int c = lane + 32 * j;
    float wb = embB[c];
    float wreg[F_];
    #pragma unroll
    for (int f = 0; f < F_; ++f) wreg[f] = embW[c * F_ + f];
    for (int r = 0; r < 16; ++r) {
      float acc = wb;
      #pragma unroll
      for (int f = 0; f < F_; ++f) acc += xs[r * F_ + f] * wreg[f];
      h0s[r * H_ + c] = (_Float16)acc;
    }
  }
  __syncthreads();

  const int g  = lane >> 4;      // half-wave group
  const int mr = lane & 15;
  const int nbase = ngrp * 128;
  v8f acc[8] = {};

  for (int kc = 0; kc < H_ / 32; ++kc) {
    const int k0 = kc * 32;
    // A fragment (16x32 f16): lane=M row; K(i) = k0 + g*16 + i
    v16h a;
    #pragma unroll
    for (int i = 0; i < 16; ++i)
      a[i] = h0s[mr * H_ + k0 + g * 16 + i];
    #pragma unroll
    for (int t = 0; t < 8; ++t) {
      const int ncol = nbase + t * 16 + mr;   // B col = output channel
      // B fragment (32x16 f16): K(i) = k0 + g*16 + i (contiguous along K)
      v16h bf;
      #pragma unroll
      for (int i = 0; i < 16; ++i)
        bf[i] = W16[(size_t)ncol * H_ + k0 + g * 16 + i];
      acc[t] = __builtin_amdgcn_wmma_f32_16x16x32_f16(
                 false, a, false, bf, (short)0, acc[t], false, false);
    }
  }
  // D layout: VGPR r -> row m0 + r + g*8, col = ncol
  #pragma unroll
  for (int t = 0; t < 8; ++t) {
    const int ncol = nbase + t * 16 + mr;
    #pragma unroll
    for (int r = 0; r < 8; ++r)
      xin[(size_t)(m0 + r + g * 8) * E_ + ncol] = acc[t][r];
  }
}

// ---------------------------------------------------------------------------
// 3) zg at last timestep only: zg[b,j] = h0[b,L-1,:] . in_proj_W[512+j,:]
// ---------------------------------------------------------------------------
__global__ __launch_bounds__(256) void k_zg_last(
    const float* __restrict__ x, const float* __restrict__ embW,
    const float* __restrict__ embB, const float* __restrict__ inW,
    float* __restrict__ zg_last)
{
  __shared__ float xrow[F_];
  __shared__ float h0row[H_];
  const int b = blockIdx.x, tid = threadIdx.x;
  const float* xr = x + ((size_t)b * L_ + (L_ - 1)) * F_;
  if (tid < F_) xrow[tid] = xr[tid];
  __syncthreads();
  {
    float acc = embB[tid];
    for (int f = 0; f < F_; ++f) acc += xrow[f] * embW[tid * F_ + f];
    h0row[tid] = acc;
  }
  __syncthreads();
  for (int jj = 0; jj < 2; ++jj) {
    const int j = tid + jj * 256;
    const float* wr = inW + (size_t)(E_ + j) * H_;
    float acc = 0.f;
    for (int k = 0; k < H_; ++k) acc += h0row[k] * wr[k];
    zg_last[b * E_ + j] = acc;
  }
}

// ---------------------------------------------------------------------------
// 4) Depthwise causal conv (K=4) + bias + SiLU -> xc (== u)
// ---------------------------------------------------------------------------
__global__ __launch_bounds__(256) void k_conv(
    const float* __restrict__ xin, const float* __restrict__ convW,
    const float* __restrict__ convB, float* __restrict__ xc)
{
  size_t idx = (size_t)blockIdx.x * 256 + threadIdx.x;
  if (idx >= (size_t)M_ * E_) return;
  const int e = (int)(idx % E_);
  const size_t m = idx / E_;
  const int l = (int)(m % L_);
  float acc = convB[e];
  #pragma unroll
  for (int k = 0; k < KC_; ++k) {
    const int dl = l + k - (KC_ - 1);
    if (dl >= 0) acc += xin[(m + k - (KC_ - 1)) * E_ + e] * convW[e * KC_ + k];
  }
  xc[idx] = acc / (1.f + __expf(-acc));   // SiLU
}

// ---------------------------------------------------------------------------
// 5) x_proj: dt (16) and B (64) columns for every token (C skipped).
// ---------------------------------------------------------------------------
__global__ __launch_bounds__(128) void k_xproj(
    const float* __restrict__ xc, const float* __restrict__ xpW,
    float* __restrict__ dtB)
{
  __shared__ float row[E_];
  const size_t m = blockIdx.x;
  for (int i = threadIdx.x; i < E_; i += 128) row[i] = xc[m * E_ + i];
  __syncthreads();
  const int j = threadIdx.x;
  if (j < DTB_) {
    const float* w = xpW + (size_t)j * E_;
    float acc = 0.f;
    for (int e = 0; e < E_; ++e) acc += row[e] * w[e];
    dtB[m * DTB_ + j] = acc;
  }
}

// ---------------------------------------------------------------------------
// 6) C at last timestep only.
// ---------------------------------------------------------------------------
__global__ __launch_bounds__(128) void k_clast(
    const float* __restrict__ xc, const float* __restrict__ xpW,
    float* __restrict__ C_last)
{
  __shared__ float row[E_];
  const int b = blockIdx.x;
  const size_t m = (size_t)b * L_ + (L_ - 1);
  for (int i = threadIdx.x; i < E_; i += 128) row[i] = xc[m * E_ + i];
  __syncthreads();
  const int n = threadIdx.x;
  if (n < N_) {
    const float* w = xpW + (size_t)(DTB_ + n) * E_;
    float acc = 0.f;
    for (int e = 0; e < E_; ++e) acc += row[e] * w[e];
    C_last[b * N_ + n] = acc;
  }
}

// ---------------------------------------------------------------------------
// 7) delta = softplus(dt @ dt_proj.T + b); prefix-sum S over l;
//    du = delta*u written in-place over xc; save xc at l=L-1.
//    Thread per (b,e): lane-contiguous in e -> coalesced.
// ---------------------------------------------------------------------------
__global__ __launch_bounds__(256) void k_prefix(
    const float* __restrict__ dtB, const float* __restrict__ dtW,
    const float* __restrict__ dtb, float* __restrict__ xc /* -> du */,
    float* __restrict__ S, float* __restrict__ xc_last)
{
  const int tid = blockIdx.x * 256 + threadIdx.x;   // 0..16383
  const int b = tid / E_, e = tid % E_;
  float w[R_];
  #pragma unroll
  for (int r = 0; r < R_; ++r) w[r] = dtW[e * R_ + r];
  const float bias = dtb[e];
  float s = 0.f;
  for (int l = 0; l < L_; ++l) {
    const size_t m = (size_t)b * L_ + l;
    const float* dt = dtB + m * DTB_;
    float pre = bias;
    #pragma unroll
    for (int r = 0; r < R_; ++r) pre += dt[r] * w[r];
    const float delta = (pre > 20.f) ? pre : log1pf(__expf(pre));  // softplus
    s += delta;
    const size_t ix = m * E_ + e;
    const float u = xc[ix];
    if (l == L_ - 1) xc_last[b * E_ + e] = u;
    xc[ix] = delta * u;   // du
    S[ix]  = s;
  }
}

// ---------------------------------------------------------------------------
// 8) Parallelized scan: h_L[b,e,n] = sum_t exp(A[e,n]*(S_L - S_t))*du_t*B_t[n]
//    One wave per (b,e); lane holds states n and n+32; shuffle-reduce with
//    C_last over n; apply +u_L*D and SiLU(zg) gate -> y[b,e].
// ---------------------------------------------------------------------------
__global__ __launch_bounds__(256) void k_scan(
    const float* __restrict__ S, const float* __restrict__ du,
    const float* __restrict__ dtB, const float* __restrict__ A_log,
    const float* __restrict__ C_last, const float* __restrict__ Dvec,
    const float* __restrict__ xc_last, const float* __restrict__ zg_last,
    float* __restrict__ y)
{
  const int wave = (blockIdx.x * 256 + threadIdx.x) >> 5;  // 0..16383
  const int lane = threadIdx.x & 31;
  const int b = wave / E_, e = wave % E_;
  const int n0 = lane, n1 = lane + 32;
  const float A0 = -__expf(A_log[e * N_ + n0]);
  const float A1 = -__expf(A_log[e * N_ + n1]);
  const float SL = S[((size_t)b * L_ + (L_ - 1)) * E_ + e];
  const float* Sp = S   + (size_t)b * L_ * E_ + e;
  const float* dp = du  + (size_t)b * L_ * E_ + e;
  const float* Bp = dtB + (size_t)b * L_ * DTB_ + R_;
  float acc0 = 0.f, acc1 = 0.f;
  for (int t = 0; t < L_; ++t) {
    const float s  = Sp[(size_t)t * E_];     // wave-uniform
    const float d  = dp[(size_t)t * E_];     // wave-uniform
    const float wdt = SL - s;                // >= 0
    const float b0 = Bp[(size_t)t * DTB_ + n0];
    const float b1 = Bp[(size_t)t * DTB_ + n1];
    acc0 += __expf(A0 * wdt) * d * b0;
    acc1 += __expf(A1 * wdt) * d * b1;
  }
  float part = acc0 * C_last[b * N_ + n0] + acc1 * C_last[b * N_ + n1];
  #pragma unroll
  for (int off = 16; off > 0; off >>= 1) part += __shfl_xor(part, off, 32);
  if (lane == 0) {
    float v = part + xc_last[b * E_ + e] * Dvec[e];
    const float z = zg_last[b * E_ + e];
    v *= z / (1.f + __expf(-z));             // * SiLU(zg)
    y[b * E_ + e] = v;
  }
}

// ---------------------------------------------------------------------------
// 9) Heads: mo = y @ out_proj.T ; feat = mo @ fc.T + b ; direction/price.
// ---------------------------------------------------------------------------
__global__ __launch_bounds__(256) void k_head(
    const float* __restrict__ y, const float* __restrict__ opW,
    const float* __restrict__ fcW, const float* __restrict__ fcB,
    const float* __restrict__ dW1, const float* __restrict__ db1,
    const float* __restrict__ dW2, const float* __restrict__ db2,
    const float* __restrict__ pW1, const float* __restrict__ pb1,
    const float* __restrict__ pW2, const float* __restrict__ pb2,
    float* __restrict__ out)
{
  __shared__ float yrow[E_];
  __shared__ float mo[H_];
  __shared__ float feat[H_];
  __shared__ float dh[64];
  __shared__ float ph[64];
  const int b = blockIdx.x, tid = threadIdx.x;
  for (int i = tid; i < E_; i += 256) yrow[i] = y[b * E_ + i];
  __syncthreads();
  { float acc = 0.f; const float* w = opW + (size_t)tid * E_;
    for (int e = 0; e < E_; ++e) acc += yrow[e] * w[e];
    mo[tid] = acc; }
  __syncthreads();
  { float acc = fcB[tid]; const float* w = fcW + (size_t)tid * H_;
    for (int k = 0; k < H_; ++k) acc += mo[k] * w[k];
    feat[tid] = acc; }
  __syncthreads();
  if (tid < 64) {
    float acc = db1[tid]; const float* w = dW1 + tid * H_;
    for (int k = 0; k < H_; ++k) acc += feat[k] * w[k];
    dh[tid] = acc > 0.f ? acc : 0.f;
  } else if (tid < 128) {
    const int j = tid - 64;
    float acc = pb1[j]; const float* w = pW1 + j * H_;
    for (int k = 0; k < H_; ++k) acc += feat[k] * w[k];
    ph[j] = acc > 0.f ? acc : 0.f;
  }
  __syncthreads();
  if (tid == 0) {
    float l0 = db2[0], l1 = db2[1];
    for (int j = 0; j < 64; ++j) { l0 += dh[j] * dW2[j]; l1 += dh[j] * dW2[64 + j]; }
    const float mx = fmaxf(l0, l1);
    const float e0 = __expf(l0 - mx), e1 = __expf(l1 - mx), sm = e0 + e1;
    out[b * 2]     = e0 / sm;
    out[b * 2 + 1] = e1 / sm;
    float p = pb2[0];
    for (int j = 0; j < 64; ++j) p += ph[j] * pW2[j];
    out[B_ * 2 + b] = p;
  }
}

// ---------------------------------------------------------------------------
extern "C" void kernel_launch(void* const* d_in, const int* in_sizes, int n_in,
                              void* d_out, int out_size, void* d_ws, size_t ws_size,
                              hipStream_t stream) {
  const float* x       = (const float*)d_in[0];
  const float* embW    = (const float*)d_in[1];
  const float* embB    = (const float*)d_in[2];
  const float* inW     = (const float*)d_in[3];
  const float* convW   = (const float*)d_in[4];
  const float* convB   = (const float*)d_in[5];
  const float* xpW     = (const float*)d_in[6];
  const float* dtW     = (const float*)d_in[7];
  const float* dtb     = (const float*)d_in[8];
  const float* A_log   = (const float*)d_in[9];
  const float* Dvec    = (const float*)d_in[10];
  const float* opW     = (const float*)d_in[11];
  const float* fcW     = (const float*)d_in[12];
  const float* fcB     = (const float*)d_in[13];
  const float* dW1     = (const float*)d_in[14];
  const float* db1     = (const float*)d_in[15];
  const float* dW2     = (const float*)d_in[16];
  const float* db2     = (const float*)d_in[17];
  const float* pW1     = (const float*)d_in[18];
  const float* pb1     = (const float*)d_in[19];
  const float* pW2     = (const float*)d_in[20];
  const float* pb2     = (const float*)d_in[21];
  float* out = (float*)d_out;

  // workspace carve-up (xin region is later reused for the prefix sums S)
  char* ws = (char*)d_ws;
  size_t off = 0;
  _Float16* W16 = (_Float16*)(ws + off); off += (size_t)E_ * H_ * sizeof(_Float16);
  off = (off + 255) & ~(size_t)255;
  float* xin_S   = (float*)(ws + off); off += (size_t)M_ * E_ * sizeof(float);   // xin -> S
  float* xc_du   = (float*)(ws + off); off += (size_t)M_ * E_ * sizeof(float);   // xc -> du
  float* dtB     = (float*)(ws + off); off += (size_t)M_ * DTB_ * sizeof(float);
  float* zg_last = (float*)(ws + off); off += (size_t)B_ * E_ * sizeof(float);
  float* xc_last = (float*)(ws + off); off += (size_t)B_ * E_ * sizeof(float);
  float* C_last  = (float*)(ws + off); off += (size_t)B_ * N_ * sizeof(float);
  float* ybuf    = (float*)(ws + off); off += (size_t)B_ * E_ * sizeof(float);
  (void)ws_size; (void)in_sizes; (void)n_in; (void)out_size;

  k_w16<<<(E_ * H_) / 256, 256, 0, stream>>>(inW, W16);
  k_embed_inproj<<<(M_ / 16) * 4, 32, 0, stream>>>(x, embW, embB, W16, xin_S);
  k_zg_last<<<B_, 256, 0, stream>>>(x, embW, embB, inW, zg_last);
  k_conv<<<(int)(((size_t)M_ * E_) / 256), 256, 0, stream>>>(xin_S, convW, convB, xc_du);
  k_xproj<<<M_, 128, 0, stream>>>(xc_du, xpW, dtB);
  k_clast<<<B_, 128, 0, stream>>>(xc_du, xpW, C_last);
  k_prefix<<<(B_ * E_) / 256, 256, 0, stream>>>(dtB, dtW, dtb, xc_du, xin_S, xc_last);
  k_scan<<<(B_ * E_ * 32) / 256, 256, 0, stream>>>(xin_S, xc_du, dtB, A_log, C_last,
                                                   Dvec, xc_last, zg_last, ybuf);
  k_head<<<B_, 256, 0, stream>>>(ybuf, opW, fcW, fcB, dW1, db1, dW2, db2,
                                 pW1, pb1, pW2, pb2, out);
}
